// RaptorXModel_36369783063180
// MI455X (gfx1250) — matrix operand order repaired and checked by reading, not compiled, not run on hardware
//
#include <hip/hip_runtime.h>
#include <hip/hip_bf16.h>

// ---------------- constants from the reference ----------------
#define LL    320           // sequence length
#define LP    322           // padded (1-px halo)
#define C1N   60            // 1D channels
#define C2N   60            // 2D channels
#define NB1   3
#define NB2   10
#define K1N   17
#define ZC_REAL 185         // 5 + 3*60
#define ZC_PAD  192         // padded to 6 slices of 32
#define GC_PAD  64          // 60 -> 64, 2 slices of 32

typedef __attribute__((ext_vector_type(16))) __bf16 v16bf;
typedef __attribute__((ext_vector_type(8)))  float  v8f;

// bf16 <-> f32 via bit manipulation (RNE), storage type = unsigned short
__device__ __forceinline__ unsigned short f2bf(float f) {
    unsigned u = __float_as_uint(f);
    unsigned r = u + 0x7FFFu + ((u >> 16) & 1u);
    return (unsigned short)(r >> 16);
}
__device__ __forceinline__ float bf2f(unsigned short s) {
    return __uint_as_float(((unsigned)s) << 16);
}

// ---------------- 1D conv (tiny, fp32 VALU) ----------------
// in:[LL][cin], w:[K1N][cin][cout], bias:[cout], optional residual add, ReLU
__global__ void k_conv1d(const float* __restrict__ in, int cin,
                         const float* __restrict__ w, const float* __restrict__ bias,
                         int cout, const float* __restrict__ resid,
                         float* __restrict__ out, int do_relu) {
    int tid = blockIdx.x * blockDim.x + threadIdx.x;
    if (tid >= LL * cout) return;
    int pos = tid / cout;
    int co  = tid % cout;
    float s = bias[co];
    for (int kk = 0; kk < K1N; ++kk) {
        int p = pos + kk - (K1N / 2);
        if (p < 0 || p >= LL) continue;
        const float* ip = in + p * cin;
        const float* wp = w + kk * cin * cout + co;
        for (int ci = 0; ci < cin; ++ci) s += ip[ci] * wp[ci * cout];
    }
    if (resid) s += resid[pos * cout + co];
    if (do_relu) s = fmaxf(s, 0.0f);
    out[pos * cout + co] = s;
}

// ---------------- zero-fill (bf16 buffers incl. halo) ----------------
__global__ void k_fill0(unsigned short* __restrict__ p, int n) {
    int tid = blockIdx.x * blockDim.x + threadIdx.x;
    if (tid < n) p[tid] = 0;
}

// ---------------- build z = concat(x2d, f_i, f_j, f_mid), bf16, halo ----------------
__global__ void k_build_z(const float* __restrict__ x2d, const float* __restrict__ h,
                          unsigned short* __restrict__ z) {
    int tid = blockIdx.x * blockDim.x + threadIdx.x;
    if (tid >= LP * LP * ZC_PAD) return;
    int c   = tid % ZC_PAD;
    int pix = tid / ZC_PAD;
    int j   = pix % LP;
    int i   = pix / LP;
    float v = 0.0f;
    if (i > 0 && i < LP - 1 && j > 0 && j < LP - 1) {
        int y = i - 1, x = j - 1;
        if (c < 5)        v = x2d[(y * LL + x) * 5 + c];
        else if (c < 65)  v = h[y * C1N + (c - 5)];
        else if (c < 125) v = h[x * C1N + (c - 65)];
        else if (c < 185) v = h[((y + x) >> 1) * C1N + (c - 125)];
    }
    z[tid] = f2bf(v);
}

// ---------------- weight repack into per-lane WMMA B layout ----------------
// w:[3][3][cin_real][cout_real] -> out blocks [tap][slice][ntile][lane 0..31][idx 0..15]
// B element (k,n) within a 32x16 slice maps to lane = (k&16)|n, idx = k&15.
__global__ void k_pack_w(const float* __restrict__ w, int cin_real, int nslices,
                         int cout_real, unsigned short* __restrict__ out, int total) {
    int tid = blockIdx.x * blockDim.x + threadIdx.x;
    if (tid >= total) return;
    int idx   = tid & 15;
    int lane  = (tid >> 4) & 31;
    int blk   = tid >> 9;
    int ntile = blk & 3;
    int s     = (blk >> 2) % nslices;
    int t     = blk / (4 * nslices);
    int k  = idx + ((lane >> 4) << 4);   // 0..31
    int n  = lane & 15;
    int ci = s * 32 + k;
    int co = ntile * 16 + n;
    float v = (ci < cin_real && co < cout_real)
                  ? w[(t * cin_real + ci) * cout_real + co] : 0.0f;
    out[tid] = f2bf(v);
}

// ---------------- 3x3 conv as implicit GEMM via v_wmma_f32_16x16x32_bf16 -----
// in : bf16 [LP][LP][cin_pad], halo zeroed. NSL = cin_pad/32 (6 for z, 2 for g)
// wpk: packed B blocks [9][NSL][4][512]. out: bf16 [LP][LP][64].
// One wave = 16 px x 64 out-ch; 8 waves/block share weights staged in LDS with
// per-tap double buffering. Per slice: clause all 8 B ds_loads, then issue the
// 4 WMMAs back-to-back (one dscnt wait instead of four).
template <int NSL>
__global__ __launch_bounds__(256)
void k_conv2d_wmma(const unsigned short* __restrict__ in, int cin_pad,
                   const unsigned short* __restrict__ wpk,
                   const float* __restrict__ bias, int cout_real,
                   const unsigned short* __restrict__ resid,
                   unsigned short* __restrict__ out, int do_relu) {
    // per-tap packed-weight block = NSL*4 ntiles * 512 halfs
    __shared__ unsigned short ldsw[2][NSL * 2048];

    const int tid  = threadIdx.x;
    const int lane = tid & 31;
    const int wib  = tid >> 5;
    const int gw   = blockIdx.x * 8 + wib;      // 6400 waves total
    const int y    = gw / (LL / 16);
    const int x0   = (gw % (LL / 16)) * 16;
    const int m    = lane & 15;                  // A row (pixel) / D column (N)
    const int hsel = lane >> 4;                  // K-chunk selector

    // stage tap-0 weights into LDS buffer 0 (256 threads, NSL uint4 each)
    {
        const uint4* src = (const uint4*)wpk;
        uint4* dst = (uint4*)(&ldsw[0][0]);
        #pragma unroll
        for (int i = 0; i < NSL; ++i) dst[tid + 256 * i] = src[tid + 256 * i];
    }
    __syncthreads();

    v8f acc[4];
    #pragma unroll
    for (int n = 0; n < 4; ++n) acc[n] = {};

    #pragma unroll
    for (int t = 0; t < 9; ++t) {
        // kick off staging of next tap's weights into the other LDS buffer
        if (t + 1 < 9) {
            const uint4* src = (const uint4*)(wpk + (size_t)(t + 1) * NSL * 2048);
            uint4* dst = (uint4*)(&ldsw[(t + 1) & 1][0]);
            #pragma unroll
            for (int i = 0; i < NSL; ++i) dst[tid + 256 * i] = src[tid + 256 * i];
            // prefetch next tap's A row for this lane (gfx1250 global_prefetch)
            const int dy2 = (t + 1) / 3 - 1;
            const int dx2 = (t + 1) % 3 - 1;
            __builtin_prefetch(in + ((y + 1 + dy2) * LP + (x0 + 1 + dx2 + m)) * cin_pad,
                               0, 1);
        }

        const int dy = t / 3 - 1;
        const int dx = t % 3 - 1;
        const unsigned short* rowp =
            in + ((y + 1 + dy) * LP + (x0 + 1 + dx + m)) * cin_pad;
        const unsigned short* wt = &ldsw[t & 1][0];

        #pragma unroll
        for (int s = 0; s < NSL; ++s) {
            union { v16bf v; uint4 q[2]; } a;
            a.q[0] = *(const uint4*)(rowp + s * 32 + 16 * hsel);
            a.q[1] = *(const uint4*)(rowp + s * 32 + 16 * hsel + 8);
            // gather all 4 N-tile B operands first (8 ds_load_b128, one wait)
            union { v16bf v; uint4 q[2]; } b[4];
            #pragma unroll
            for (int n = 0; n < 4; ++n) {
                const unsigned short* wb = wt + (s * 4 + n) * 512 + lane * 16;
                b[n].q[0] = *(const uint4*)(wb);
                b[n].q[1] = *(const uint4*)(wb + 8);
            }
            // then the 4 WMMAs back-to-back
            #pragma unroll
            for (int n = 0; n < 4; ++n) {
                acc[n] = __builtin_amdgcn_wmma_f32_16x16x32_bf16(
                    false, a.v, false, b[n].v, (short)0, acc[n], false, false);
            }
        }
        __syncthreads();   // staging done before next compute; compute done before reuse
    }

    // epilogue: D layout -> lane&15 = N, vgpr r holds M = r + 8*hsel
    #pragma unroll
    for (int n = 0; n < 4; ++n) {
        const int ch = n * 16 + m;
        const float bv = (ch < cout_real) ? bias[ch] : 0.0f;
        #pragma unroll
        for (int r = 0; r < 8; ++r) {
            const int px  = x0 + r + 8 * hsel;
            const int idx = ((y + 1) * LP + (px + 1)) * GC_PAD + ch;
            float v = acc[n][r] + bv;
            if (resid) v += bf2f(resid[idx]);
            if (do_relu) v = fmaxf(v, 0.0f);
            out[idx] = f2bf(v);
        }
    }
}

// ---------------- FC head: 60 -> 100 (ReLU) -> 2, fp32 ----------------
__global__ void k_fc(const unsigned short* __restrict__ g,
                     const float* __restrict__ wfc1, const float* __restrict__ bfc1,
                     const float* __restrict__ wfc2, const float* __restrict__ bfc2,
                     float* __restrict__ out) {
    int tid = blockIdx.x * blockDim.x + threadIdx.x;
    if (tid >= LL * LL) return;
    int y = tid / LL, x = tid % LL;
    const unsigned short* gp = g + ((y + 1) * LP + (x + 1)) * GC_PAD;
    float gv[C2N];
    #pragma unroll
    for (int c = 0; c < C2N; ++c) gv[c] = bf2f(gp[c]);
    float o0 = bfc2[0], o1 = bfc2[1];
    for (int d = 0; d < 100; ++d) {
        float s = bfc1[d];
        #pragma unroll
        for (int c = 0; c < C2N; ++c) s += gv[c] * wfc1[c * 100 + d];
        s = fmaxf(s, 0.0f);
        o0 += s * wfc2[d * 2 + 0];
        o1 += s * wfc2[d * 2 + 1];
    }
    out[tid * 2 + 0] = o0;
    out[tid * 2 + 1] = o1;
}

// ---------------- host-side orchestration ----------------
extern "C" void kernel_launch(void* const* d_in, const int* in_sizes, int n_in,
                              void* d_out, int out_size, void* d_ws, size_t ws_size,
                              hipStream_t stream) {
    const float* x1d    = (const float*)d_in[0];
    const float* x2d    = (const float*)d_in[1];
    const float* w1d_in = (const float*)d_in[2];
    const float* b1d_in = (const float*)d_in[3];
    const float* w1d_a  = (const float*)d_in[4];
    const float* b1d_a  = (const float*)d_in[5];
    const float* w1d_b  = (const float*)d_in[6];
    const float* b1d_b  = (const float*)d_in[7];
    const float* w2d_in = (const float*)d_in[8];
    const float* b2d_in = (const float*)d_in[9];
    const float* w2d_a  = (const float*)d_in[10];
    const float* b2d_a  = (const float*)d_in[11];
    const float* w2d_b  = (const float*)d_in[12];
    const float* b2d_b  = (const float*)d_in[13];
    const float* wfc1   = (const float*)d_in[14];
    const float* bfc1   = (const float*)d_in[15];
    const float* wfc2   = (const float*)d_in[16];
    const float* bfc2   = (const float*)d_in[17];
    float* outp = (float*)d_out;

    char* ws = (char*)d_ws;
    // workspace layout (all offsets multiple of 256B)
    const size_t off_h  = 0;                                   // [320][60] f32
    const size_t off_t1 = 76800;                               // [320][60] f32
    const size_t off_z  = 153600;                              // [322][322][192] bf16
    const size_t sz_z   = (size_t)LP * LP * ZC_PAD * 2;        // 39,814,656
    const size_t off_g  = off_z + sz_z;                        // [322][322][64] bf16
    const size_t sz_g   = (size_t)LP * LP * GC_PAD * 2;        // 13,271,552
    const size_t off_t2 = off_g + sz_g;
    const size_t off_wz = off_t2 + sz_g;                       // packed w2d_in
    const size_t sz_wz  = (size_t)9 * 6 * 4 * 512 * 2;         // 221,184
    const size_t off_wa = off_wz + sz_wz;                      // 10 packed blocks a
    const size_t sz_wc  = (size_t)9 * 2 * 4 * 512 * 2;         // 73,728 per conv
    const size_t off_wb = off_wa + 10 * sz_wc;                 // 10 packed blocks b

    float* h  = (float*)(ws + off_h);
    float* t1 = (float*)(ws + off_t1);
    unsigned short* z  = (unsigned short*)(ws + off_z);
    unsigned short* g  = (unsigned short*)(ws + off_g);
    unsigned short* t2 = (unsigned short*)(ws + off_t2);
    unsigned short* wz = (unsigned short*)(ws + off_wz);

    const int TPB = 256;

    // ---- 1D tower ----
    {
        int n = LL * C1N, blk = (n + TPB - 1) / TPB;
        k_conv1d<<<blk, TPB, 0, stream>>>(x1d, 26, w1d_in, b1d_in, C1N, nullptr, h, 1);
        for (int k = 0; k < NB1; ++k) {
            k_conv1d<<<blk, TPB, 0, stream>>>(h, C1N, w1d_a + (size_t)k * K1N * C1N * C1N,
                                              b1d_a + k * C1N, C1N, nullptr, t1, 1);
            k_conv1d<<<blk, TPB, 0, stream>>>(t1, C1N, w1d_b + (size_t)k * K1N * C1N * C1N,
                                              b1d_b + k * C1N, C1N, /*resid=*/h, h, 1);
        }
    }

    // ---- pairwise feature tensor z (bf16, zero halo) ----
    {
        int n = LP * LP * ZC_PAD;
        k_build_z<<<(n + TPB - 1) / TPB, TPB, 0, stream>>>(x2d, h, z);
    }

    // ---- zero g/t2 (halo must be 0; interior overwritten anyway) ----
    {
        int n = LP * LP * GC_PAD;
        k_fill0<<<(n + TPB - 1) / TPB, TPB, 0, stream>>>(g, n);
        k_fill0<<<(n + TPB - 1) / TPB, TPB, 0, stream>>>(t2, n);
    }

    // ---- repack conv2d weights to WMMA B layout ----
    {
        int tin = 9 * 6 * 4 * 512;
        k_pack_w<<<(tin + TPB - 1) / TPB, TPB, 0, stream>>>(w2d_in, ZC_REAL, 6, C2N, wz, tin);
        int tc = 9 * 2 * 4 * 512;
        for (int k = 0; k < NB2; ++k) {
            k_pack_w<<<(tc + TPB - 1) / TPB, TPB, 0, stream>>>(
                w2d_a + (size_t)k * 9 * C2N * C2N, C2N, 2, C2N,
                (unsigned short*)(ws + off_wa + k * sz_wc), tc);
            k_pack_w<<<(tc + TPB - 1) / TPB, TPB, 0, stream>>>(
                w2d_b + (size_t)k * 9 * C2N * C2N, C2N, 2, C2N,
                (unsigned short*)(ws + off_wb + k * sz_wc), tc);
        }
    }

    // ---- 2D tower: WMMA implicit-GEMM convs, weights staged in LDS ----
    const int convBlocks = (LL / 16) * LL / 8;   // 6400 waves / 8 per block = 800
    k_conv2d_wmma<6><<<convBlocks, TPB, 0, stream>>>(z, ZC_PAD, wz, b2d_in, C2N,
                                                     nullptr, g, 1);
    for (int k = 0; k < NB2; ++k) {
        const unsigned short* wa = (const unsigned short*)(ws + off_wa + k * sz_wc);
        const unsigned short* wb = (const unsigned short*)(ws + off_wb + k * sz_wc);
        k_conv2d_wmma<2><<<convBlocks, TPB, 0, stream>>>(g, GC_PAD, wa, b2d_a + k * C2N,
                                                         C2N, nullptr, t2, 1);
        k_conv2d_wmma<2><<<convBlocks, TPB, 0, stream>>>(t2, GC_PAD, wb, b2d_b + k * C2N,
                                                         C2N, /*resid=*/g, g, 1);
    }

    // ---- FC head ----
    {
        int n = LL * LL;
        k_fc<<<(n + TPB - 1) / TPB, TPB, 0, stream>>>(g, wfc1, bfc1, wfc2, bfc2, outp);
    }
}
